// LlamaAttention_heavy_hitter_45406394253461
// MI455X (gfx1250) — matrix operand order repaired and checked by reading, not compiled, not run on hardware
//
#include <hip/hip_runtime.h>
#include <hip/hip_bf16.h>

#define HIDDEN 4096
#define NH     32
#define NKV    8
#define HD     128
#define SEQ    2048
#define GROUPS (NH / NKV)
#define NCHUNK (SEQ / 16)
#define KTOP   8
#define INIT_B 128
#define REC_B  128

typedef __attribute__((ext_vector_type(16))) _Float16 v16h;
typedef __attribute__((ext_vector_type(8)))  _Float16 v8h;
typedef __attribute__((ext_vector_type(8)))  float    v8f;

#define NEGF  (-3.4028234663852886e38f)
#define SCALE 0.08838834764831845f   // 1/sqrt(128)

// ---------------------------------------------------------------- casts
__global__ __launch_bounds__(256) void cast_f32_to_f16(
    const float* __restrict__ src, _Float16* __restrict__ dst, long long n)
{
    long long i = (long long)blockIdx.x * blockDim.x + threadIdx.x;
    const long long stride = (long long)gridDim.x * blockDim.x;
    for (; i < n; i += stride) dst[i] = (_Float16)src[i];
}

// A fragment (16x32 f16): lane half 0 holds K {0..15}, half 1 {16..31}
__device__ __forceinline__ v16h load_afrag(const _Float16* p)
{
    v8h lo = *(const v8h*)(p);
    v8h hi = *(const v8h*)(p + 8);
    v16h a;
#pragma unroll
    for (int j = 0; j < 8; ++j) { a[j] = lo[j]; a[j + 8] = hi[j]; }
    return a;
}

// ---------------------------------------------------------------- GEMM (NT)
// C[M,N] (f32) = A[M,K] (f16, row-major) * B[N,K]^T (f16, row-major)
// grid = (N/256, M/128), block = 256 (8 waves in 2x4); wave computes 64x64
// via 4x4 accumulators: 16 WMMAs per 32-wide k-step, 8 fragment loads.
__global__ __launch_bounds__(256) void wmma_gemm_nt(
    const _Float16* __restrict__ A, const _Float16* __restrict__ B,
    float* __restrict__ C, int M, int N, int K)
{
    const int lane = threadIdx.x & 31;
    const int wv   = threadIdx.x >> 5;
    const int lm   = lane & 15;
    const int hf   = lane >> 4;
    const int m0   = blockIdx.y * 128 + (wv >> 2) * 64;
    const int n0   = blockIdx.x * 256 + (wv & 3) * 64;

    const _Float16* ar[4];
    const _Float16* br[4];
#pragma unroll
    for (int mi = 0; mi < 4; ++mi)
        ar[mi] = A + (size_t)(m0 + mi * 16 + lm) * K + hf * 16;
#pragma unroll
    for (int nj = 0; nj < 4; ++nj)
        br[nj] = B + (size_t)(n0 + nj * 16 + lm) * K + hf * 16;

    v8f acc[4][4] = {};
    for (int kk = 0; kk < K; kk += 32) {
        v16h a[4], b[4];
#pragma unroll
        for (int mi = 0; mi < 4; ++mi) a[mi] = load_afrag(ar[mi] + kk);
#pragma unroll
        for (int nj = 0; nj < 4; ++nj) b[nj] = *(const v16h*)(br[nj] + kk);
#pragma unroll
        for (int mi = 0; mi < 4; ++mi)
#pragma unroll
            for (int nj = 0; nj < 4; ++nj)
                acc[mi][nj] = __builtin_amdgcn_wmma_f32_16x16x32_f16(
                    false, a[mi], false, b[nj], (short)0, acc[mi][nj], false, false);
    }
#pragma unroll
    for (int mi = 0; mi < 4; ++mi)
#pragma unroll
        for (int i = 0; i < 8; ++i) {
            const int row = m0 + mi * 16 + hf * 8 + i;
            float* crow = C + (size_t)row * N + n0 + lm;
#pragma unroll
            for (int nj = 0; nj < 4; ++nj)
                crow[nj * 16] = acc[mi][nj][i];
        }
}

// ---------------------------------------------------------------- RoPE + head split
// grid = (SEQ, NH + 2*NKV), block = 128.
// Q -> Qh[h][s][d] (row-major per head), K -> Kh[kv][s][d],
// V -> Vt[kv][d][s]  (TRANSPOSED so PV B-fragments become contiguous loads)
__global__ __launch_bounds__(128) void rope_split_heads(
    const float* __restrict__ Qf, const float* __restrict__ Kf, const float* __restrict__ Vf,
    _Float16* __restrict__ Qh, _Float16* __restrict__ Kh, _Float16* __restrict__ Vt)
{
    const int s = blockIdx.x;
    const int y = blockIdx.y;
    const int d = threadIdx.x;
    const int j = d & 63;
    const float ang = (float)s * powf(10000.0f, -(float)j * (1.0f / 64.0f));
    const float c = cosf(ang), sn = sinf(ang);
    if (y < NH) {
        const float* src = Qf + (size_t)s * (NH * HD) + y * HD;
        const float x  = src[d];
        const float xr = (d < 64) ? -src[d + 64] : src[d - 64];
        Qh[((size_t)y * SEQ + s) * HD + d] = (_Float16)(x * c + xr * sn);
    } else if (y < NH + NKV) {
        const int hk = y - NH;
        const float* src = Kf + (size_t)s * (NKV * HD) + hk * HD;
        const float x  = src[d];
        const float xr = (d < 64) ? -src[d + 64] : src[d - 64];
        Kh[((size_t)hk * SEQ + s) * HD + d] = (_Float16)(x * c + xr * sn);
    } else {
        const int hk = y - NH - NKV;
        Vt[((size_t)hk * HD + d) * SEQ + s] =
            (_Float16)Vf[(size_t)s * (NKV * HD) + hk * HD + d];
    }
}

// ---------------------------------------------------------------- attention helpers
__device__ __forceinline__ float warp_max16(float v) {
    v = fmaxf(v, __shfl_xor(v, 1, 32));
    v = fmaxf(v, __shfl_xor(v, 2, 32));
    v = fmaxf(v, __shfl_xor(v, 4, 32));
    v = fmaxf(v, __shfl_xor(v, 8, 32));
    return v;
}
__device__ __forceinline__ float warp_sum16(float v) {
    v += __shfl_xor(v, 1, 32);
    v += __shfl_xor(v, 2, 32);
    v += __shfl_xor(v, 4, 32);
    v += __shfl_xor(v, 8, 32);
    return v;
}

// 16x16 score tile = Q(16x128) x K_tile(16x128)^T, K accumulated over 4 WMMAs
__device__ __forceinline__ v8f score_tile(const v16h aQ[4], const _Float16* kbase,
                                          int lm, int hf)
{
    v8f acc = {};
#pragma unroll
    for (int st = 0; st < 4; ++st) {
        v16h b = *(const v16h*)(kbase + (size_t)lm * HD + st * 32 + hf * 16);
        acc = __builtin_amdgcn_wmma_f32_16x16x32_f16(false, aQ[st], false, b, (short)0, acc, false, false);
    }
    return acc;
}

// ---------------------------------------------------------------- Quest attention
// grid = (SEQ/16, NH), block = 32 (one wave per (head, 16-query tile))
__global__ __launch_bounds__(32) void quest_attn(
    const _Float16* __restrict__ Qh, const _Float16* __restrict__ Kh,
    const _Float16* __restrict__ Vt, _Float16* __restrict__ Ao)
{
    __shared__ float    cmax[16][NCHUNK];   // per-row per-chunk max
    __shared__ unsigned sel[16][4];         // per-row 128-bit top-8 chunk mask
    __shared__ _Float16 pt[16][40];         // P tile staging (C-layout -> A-layout)

    const int lane = threadIdx.x;
    const int lm = lane & 15;
    const int hf = lane >> 4;
    const int qt = blockIdx.x;
    const int h  = blockIdx.y;
    const int kvh = h / GROUPS;
    const int q0 = qt * 16;

    // Load Q fragments once (4 K-steps of 32)
    v16h aQ[4];
    {
        const _Float16* qrow = Qh + ((size_t)h * SEQ + q0 + lm) * HD;
#pragma unroll
        for (int st = 0; st < 4; ++st)
            aQ[st] = load_afrag(qrow + st * 32 + hf * 16);
    }
    const _Float16* Kbase  = Kh + (size_t)kvh * SEQ * HD;
    const _Float16* Vtbase = Vt + (size_t)kvh * HD * SEQ;

    // ---- pass 1: chunk maxima of causal-masked scaled scores (chunk == 16-key tile)
    for (int c = 0; c <= qt; ++c) {
        v8f sc = score_tile(aQ, Kbase + (size_t)c * 16 * HD, lm, hf);
#pragma unroll
        for (int i = 0; i < 8; ++i) {
            const int row = q0 + hf * 8 + i;
            const int key = c * 16 + lm;
            float v = (key <= row) ? sc[i] * SCALE : NEGF;
            v = warp_max16(v);
            if (lm == 0) cmax[hf * 8 + i][c] = v;
        }
    }
    __syncthreads();

    // ---- exact top-8 chunk selection per query row
    if (lane < 16) {
        unsigned w0 = 0, w1 = 0, w2 = 0, w3 = 0;
        const int nch = qt + 1;
        for (int t = 0; t < KTOP; ++t) {
            float best = -__builtin_inff();
            int bi = 0;
            for (int c = 0; c < nch; ++c) {
                const float v = cmax[lane][c];
                if (v > best) { best = v; bi = c; }
            }
            cmax[lane][bi] = -__builtin_inff();
            if (bi < 32)       w0 |= 1u << bi;
            else if (bi < 64)  w1 |= 1u << (bi - 32);
            else if (bi < 96)  w2 |= 1u << (bi - 64);
            else               w3 |= 1u << (bi - 96);
        }
        sel[lane][0] = w0; sel[lane][1] = w1; sel[lane][2] = w2; sel[lane][3] = w3;
    }
    __syncthreads();

    // ---- pass 2: flash-style masked softmax + PV over 32-key double tiles
    float m_i[8], l_i[8];
    v8f o[8] = {};
#pragma unroll
    for (int i = 0; i < 8; ++i) { m_i[i] = NEGF; l_i[i] = 0.0f; }

    const int ndt = (qt + 2) >> 1;          // ceil((qt+1)/2)
    for (int dt = 0; dt < ndt; ++dt) {
        const int kb = dt * 32;
        v8f s0 = score_tile(aQ, Kbase + (size_t)kb * HD, lm, hf);
        v8f s1 = score_tile(aQ, Kbase + (size_t)(kb + 16) * HD, lm, hf);

        float p0[8], p1[8], fac[8];
#pragma unroll
        for (int i = 0; i < 8; ++i) {
            const int rloc = hf * 8 + i;
            const int row  = q0 + rloc;
            const int k0 = kb + lm;
            const int k1 = kb + 16 + lm;
            const unsigned* sw = sel[rloc];
            float v0 = s0[i] * SCALE;
            float v1 = s1[i] * SCALE;
            const bool a0 = (k0 <= row) && ((k0 < INIT_B) || (k0 > row - REC_B) ||
                            ((sw[(k0 >> 4) >> 5] >> ((k0 >> 4) & 31)) & 1u));
            const bool a1 = (k1 <= row) && ((k1 < INIT_B) || (k1 > row - REC_B) ||
                            ((sw[(k1 >> 4) >> 5] >> ((k1 >> 4) & 31)) & 1u));
            v0 = a0 ? v0 : NEGF;
            v1 = a1 ? v1 : NEGF;
            const float tm   = warp_max16(fmaxf(v0, v1));
            const float mnew = fmaxf(m_i[i], tm);
            fac[i] = __expf(m_i[i] - mnew);
            m_i[i] = mnew;
            p0[i] = __expf(v0 - mnew);
            p1[i] = __expf(v1 - mnew);
            l_i[i] = l_i[i] * fac[i] + warp_sum16(p0[i] + p1[i]);
        }
#pragma unroll
        for (int t = 0; t < 8; ++t)
#pragma unroll
            for (int i = 0; i < 8; ++i)
                o[t][i] *= fac[i];

        __syncthreads();
#pragma unroll
        for (int i = 0; i < 8; ++i) {
            const int rloc = hf * 8 + i;
            pt[rloc][lm]      = (_Float16)p0[i];
            pt[rloc][16 + lm] = (_Float16)p1[i];
        }
        __syncthreads();

        // P as A-fragment (16x32)
        v16h ap;
        {
            v8h lo = *(const v8h*)(&pt[lm][hf * 16]);
            v8h hi = *(const v8h*)(&pt[lm][hf * 16 + 8]);
#pragma unroll
            for (int jj = 0; jj < 8; ++jj) { ap[jj] = lo[jj]; ap[jj + 8] = hi[jj]; }
        }
        // V as B-fragments from transposed layout: one contiguous 32B load per lane
#pragma unroll
        for (int t = 0; t < 8; ++t) {
            v16h bv = *(const v16h*)(Vtbase + (size_t)(t * 16 + lm) * SEQ + kb + hf * 16);
            o[t] = __builtin_amdgcn_wmma_f32_16x16x32_f16(false, ap, false, bv, (short)0, o[t], false, false);
        }
    }

    // ---- epilogue: normalize, write head-major features [s][h*128+d] as f16
#pragma unroll
    for (int i = 0; i < 8; ++i) {
        const float inv = 1.0f / l_i[i];
        const int row = q0 + hf * 8 + i;
        _Float16* orow = Ao + (size_t)row * HIDDEN + h * HD;
#pragma unroll
        for (int t = 0; t < 8; ++t)
            orow[t * 16 + lm] = (_Float16)(o[t][i] * inv);
    }
}

// ---------------------------------------------------------------- host
extern "C" void kernel_launch(void* const* d_in, const int* in_sizes, int n_in,
                              void* d_out, int out_size, void* d_ws, size_t ws_size,
                              hipStream_t stream)
{
    (void)in_sizes; (void)n_in; (void)out_size; (void)ws_size;
    const float* hs = (const float*)d_in[0];
    const float* wq = (const float*)d_in[1];
    const float* wk = (const float*)d_in[2];
    const float* wv = (const float*)d_in[3];
    const float* wo = (const float*)d_in[4];
    float* out = (float*)d_out;

    char* ws = (char*)d_ws;
    size_t off = 0;
    auto alloc = [&](size_t bytes) -> void* {
        void* p = ws + off;
        off = (off + bytes + 255) & ~(size_t)255;
        return p;
    };

    _Float16* Xh  = (_Float16*)alloc((size_t)SEQ * HIDDEN * 2);
    _Float16* Wqh = (_Float16*)alloc((size_t)HIDDEN * HIDDEN * 2);
    _Float16* Wkh = (_Float16*)alloc((size_t)(NKV * HD) * HIDDEN * 2);
    _Float16* Wvh = (_Float16*)alloc((size_t)(NKV * HD) * HIDDEN * 2);
    _Float16* Woh = (_Float16*)alloc((size_t)HIDDEN * HIDDEN * 2);
    float*    Qf  = (float*)alloc((size_t)SEQ * HIDDEN * 4);
    float*    Kf  = (float*)alloc((size_t)SEQ * (NKV * HD) * 4);
    float*    Vf  = (float*)alloc((size_t)SEQ * (NKV * HD) * 4);
    _Float16* Qhh = (_Float16*)alloc((size_t)NH  * SEQ * HD * 2);
    _Float16* Khh = (_Float16*)alloc((size_t)NKV * SEQ * HD * 2);
    _Float16* Vtt = (_Float16*)alloc((size_t)NKV * HD * SEQ * 2);
    _Float16* Ao  = (_Float16*)alloc((size_t)SEQ * HIDDEN * 2);

    cast_f32_to_f16<<<512, 256, 0, stream>>>(hs, Xh, (long long)SEQ * HIDDEN);
    cast_f32_to_f16<<<1024, 256, 0, stream>>>(wq, Wqh, (long long)HIDDEN * HIDDEN);
    cast_f32_to_f16<<<512, 256, 0, stream>>>(wk, Wkh, (long long)(NKV * HD) * HIDDEN);
    cast_f32_to_f16<<<512, 256, 0, stream>>>(wv, Wvh, (long long)(NKV * HD) * HIDDEN);
    cast_f32_to_f16<<<1024, 256, 0, stream>>>(wo, Woh, (long long)HIDDEN * HIDDEN);

    dim3 gq(HIDDEN / 256, SEQ / 128);
    dim3 gk((NKV * HD) / 256, SEQ / 128);
    wmma_gemm_nt<<<gq, 256, 0, stream>>>(Xh, Wqh, Qf, SEQ, HIDDEN, HIDDEN);
    wmma_gemm_nt<<<gk, 256, 0, stream>>>(Xh, Wkh, Kf, SEQ, NKV * HD, HIDDEN);
    wmma_gemm_nt<<<gk, 256, 0, stream>>>(Xh, Wvh, Vf, SEQ, NKV * HD, HIDDEN);

    rope_split_heads<<<dim3(SEQ, NH + 2 * NKV), 128, 0, stream>>>(Qf, Kf, Vf, Qhh, Khh, Vtt);

    quest_attn<<<dim3(SEQ / 16, NH), 32, 0, stream>>>(Qhh, Khh, Vtt, Ao);

    wmma_gemm_nt<<<gq, 256, 0, stream>>>(Ao, Woh, out, SEQ, HIDDEN, HIDDEN);
}
